// RNNLayer_67353677136639
// MI455X (gfx1250) — matrix operand order repaired and checked by reading, not compiled, not run on hardware
//
#include <hip/hip_runtime.h>
#include <hip/hip_bf16.h>

typedef __attribute__((ext_vector_type(16))) _Float16 v16h;
typedef __attribute__((ext_vector_type(8)))  _Float16 v8h;
typedef __attribute__((ext_vector_type(8)))  float    v8f;
typedef __attribute__((ext_vector_type(4)))  float    f32x4;

#define Tdim 2048
#define Bdim 64
#define Vdim 512
#define Hdim 512
#define Wcols (Vdim + Hdim)   // 1024
#define LDS_STRIDE 520        // 512 + 8 halves: 4-bank row offset, 16B-aligned rows

__device__ __forceinline__ float fast_tanh(float x) {
#if __has_builtin(__builtin_amdgcn_tanhf)
  return __builtin_amdgcn_tanhf(x);
#elif __has_builtin(__builtin_amdgcn_tanh_f32)
  return __builtin_amdgcn_tanh_f32(x);
#else
  float r;
  // CDNA5 TRANS op; v_nop covers the 1-op TRANS result hazard (ISA 7.4).
  asm volatile("v_tanh_f32 %0, %1\n\tv_nop" : "=v"(r) : "v"(x));
  return r;
#endif
}

// ---------------- Kernel 0: convert W (f32, HxWcols) -> f16 in d_ws ----------------
__global__ void RNN_wcvt(const float* __restrict__ W, _Float16* __restrict__ W16, int n) {
  for (int i = blockIdx.x * blockDim.x + threadIdx.x; i < n; i += gridDim.x * blockDim.x)
    W16[i] = (_Float16)W[i];
}

// ---------------- Kernel 1: Z = X @ Wx^T + b  (parallel, memory-bound) ----------------
__global__ __launch_bounds__(256) void RNN_gemm_x(const float* __restrict__ X,
                                                  const _Float16* __restrict__ W16,
                                                  const float* __restrict__ bias,
                                                  float* __restrict__ Z) {
  const int lane = threadIdx.x & 31;
  const int wid  = blockIdx.x * (blockDim.x >> 5) + (threadIdx.x >> 5);
  const int row_tile = wid >> 3;        // (T*B)/16 = 8192 row tiles
  const int cq       = wid & 7;         // 8 col-quads of 64 columns
  const int r0 = row_tile * 16;
  const int j0 = cq * 64;
  const int m  = lane & 15;
  const int hi = lane >> 4;
  const int abase = hi ? 16 : 0;        // A frag K-offset (halves)
  const int bbase = hi ? 16 : 0;        // B frag K-offset (halves)

  v8f c[4];
#pragma unroll
  for (int q = 0; q < 4; ++q) {
    const float bv = bias[j0 + q * 16 + m];
#pragma unroll
    for (int r = 0; r < 8; ++r) c[q][r] = bv;
  }

  const float* xrow = X + (size_t)(r0 + m) * Vdim;

#pragma unroll 4
  for (int kk = 0; kk < Vdim / 32; ++kk) {
    const int k0 = kk * 32;
    __builtin_prefetch(xrow + k0 + 128, 0, 1);           // global_prefetch_b8
    const f32x4 f0 = *(const f32x4*)(xrow + k0 + abase);
    const f32x4 f1 = *(const f32x4*)(xrow + k0 + abase + 4);
    const f32x4 f2 = *(const f32x4*)(xrow + k0 + abase + 8);
    const f32x4 f3 = *(const f32x4*)(xrow + k0 + abase + 12);
    v16h a;
#pragma unroll
    for (int i = 0; i < 4; ++i) {
      a[i]      = (_Float16)f0[i];
      a[4 + i]  = (_Float16)f1[i];
      a[8 + i]  = (_Float16)f2[i];
      a[12 + i] = (_Float16)f3[i];
    }
#pragma unroll
    for (int q = 0; q < 4; ++q) {
      const v16h bf = *(const v16h*)(W16 + (size_t)(j0 + q * 16 + m) * Wcols + k0 + bbase);
      c[q] = __builtin_amdgcn_wmma_f32_16x16x32_f16(false, a, false, bf, (short)0, c[q],
                                                    false, false);
    }
  }

#pragma unroll
  for (int q = 0; q < 4; ++q) {
    const int j = j0 + q * 16 + m;
#pragma unroll
    for (int r = 0; r < 8; ++r)
      Z[(size_t)(r0 + r + hi * 8) * Hdim + j] = c[q][r];
  }
}

// ---------------- Kernel 2: persistent recurrence on one WGP ----------------
// Mapping: wave w owns ONE column tile (j0 = w*16) across ALL 4 batch-row tiles.
// => each Wh fragment is loaded by exactly one wave, one B frag feeds 4 WMMAs;
//    A frags come from LDS (conflict-free 4-bank-offset stride).
// Z[t+1] is prefetched during step t (1024 threads x 128B = full 128KB tile);
// streaming Z traffic uses non-temporal hints so Wh stays L2-resident.
__global__ __launch_bounds__(1024) void RNN_recur(float* __restrict__ states,
                                                  const _Float16* __restrict__ W16,
                                                  const float* __restrict__ h0,
                                                  float* __restrict__ hfinal) {
  __shared__ _Float16 hbuf[Bdim * LDS_STRIDE];   // 66,560 B
  const int tid  = threadIdx.x;
  const int lane = tid & 31;
  const int w    = tid >> 5;            // 0..31  -> col tile
  const int j0   = w * 16;
  const int m    = lane & 15;
  const int hi   = lane >> 4;
  const int abase = hi ? 16 : 0;
  const int bbase = hi ? 16 : 0;

  // init hidden state from h0 (f32 -> f16)
  for (int i = tid; i < Bdim * Hdim; i += 1024) {
    const int bb = i >> 9, jj = i & 511;
    hbuf[bb * LDS_STRIDE + jj] = (_Float16)h0[i];
  }
  __syncthreads();

  const _Float16* Wh   = W16 + Vdim;            // Wh = cols [V, V+H) of 1024-wide rows
  const _Float16* brow = Wh + (size_t)(j0 + m) * Wcols + bbase;

  for (int t = 0; t < Tdim; ++t) {
    float* Zt = states + (size_t)t * Bdim * Hdim;

    v8f c[4];
#pragma unroll
    for (int mt = 0; mt < 4; ++mt)
#pragma unroll
      for (int r = 0; r < 8; ++r)
        c[mt][r] = __builtin_nontemporal_load(
            Zt + (size_t)(mt * 16 + r + hi * 8) * Hdim + j0 + m);

    // Hide next step's HBM latency behind this step's K-loop:
    // 1024 threads x 128B lines cover the whole 128KB Z[t+1] tile.
    if (t + 1 < Tdim)
      __builtin_prefetch((const char*)(Zt + Bdim * Hdim) + tid * 128, 0, 1);

    // B double buffer: next Wh fragment loads while 4 WMMAs consume current one
    v16h bf = *(const v16h*)(brow);
#pragma unroll 4
    for (int kk = 0; kk < Hdim / 32; ++kk) {
      const int k0 = kk * 32;
      const v16h bnext = (kk + 1 < Hdim / 32) ? *(const v16h*)(brow + k0 + 32) : bf;
#pragma unroll
      for (int mt = 0; mt < 4; ++mt) {
        const _Float16* arow = hbuf + (mt * 16 + m) * LDS_STRIDE + k0 + abase;
        const v8h lo8 = *(const v8h*)(arow);
        const v8h hi8 = *(const v8h*)(arow + 8);
        const v16h a = __builtin_shufflevector(lo8, hi8, 0, 1, 2, 3, 4, 5, 6, 7,
                                               8, 9, 10, 11, 12, 13, 14, 15);
        c[mt] = __builtin_amdgcn_wmma_f32_16x16x32_f16(false, a, false, bf, (short)0,
                                                       c[mt], false, false);
      }
      bf = bnext;
    }

    __syncthreads();   // all waves done reading hbuf for this step

#pragma unroll
    for (int mt = 0; mt < 4; ++mt)
#pragma unroll
      for (int r = 0; r < 8; ++r) {
        const int bb = mt * 16 + r + hi * 8;
        const float v = fast_tanh(c[mt][r]);
        __builtin_nontemporal_store(v, Zt + (size_t)bb * Hdim + j0 + m);  // states[t]
        hbuf[bb * LDS_STRIDE + j0 + m] = (_Float16)v;                     // h for t+1
        if (t == Tdim - 1) hfinal[(size_t)bb * Hdim + j0 + m] = v;
      }
    __syncthreads();   // hbuf fully updated before next step's reads
  }
}

extern "C" void kernel_launch(void* const* d_in, const int* in_sizes, int n_in,
                              void* d_out, int out_size, void* d_ws, size_t ws_size,
                              hipStream_t stream) {
  const float* X    = (const float*)d_in[0];   // (T, B, V)
  const float* h0   = (const float*)d_in[1];   // (1, B, H)
  const float* W    = (const float*)d_in[2];   // (H, V+H)
  const float* bias = (const float*)d_in[3];   // (H,)

  float* states = (float*)d_out;                          // (T, B, H)
  float* hfinal = states + (size_t)Tdim * Bdim * Hdim;    // (1, B, H)
  _Float16* W16 = (_Float16*)d_ws;                        // H * Wcols halves = 1 MB

  RNN_wcvt<<<256, 512, 0, stream>>>(W, W16, Hdim * Wcols);

  // (T*B/16) row tiles * 8 col-quads = 65536 wave-tasks; 8 waves/block -> 8192 blocks
  RNN_gemm_x<<<8192, 256, 0, stream>>>(X, W16, bias, states);

  RNN_recur<<<1, 1024, 0, stream>>>(states, W16, h0, hfinal);
}
